// ArcFaceLoss_36885179138676
// MI455X (gfx1250) — matrix operand order repaired and checked by reading, not compiled, not run on hardware
//
#include <hip/hip_runtime.h>
#include <hip/hip_bf16.h>
#include <math.h>

#define NUM_CLASSES 100000
#define EMBED       384
#define BATCH       512
#define NTILES      (NUM_CLASSES / 16)   // 6250 class tiles

// LDS B-tile: 16 rows x 384 bf16, padded to 392 bf16 (784 B) per row so that
// 16 lanes reading 16 different rows at the same column hit distinct banks
// (784 B = 196 dwords == 4 mod 64).
#define BROW_PAD    392

typedef __bf16 bf16_t;
typedef __attribute__((ext_vector_type(16))) __bf16 v16bf;
typedef __attribute__((ext_vector_type(8)))  __bf16 v8bf;
typedef __attribute__((ext_vector_type(8)))  float  v8f;

// ArcFace constants (m = 0.5, s = 64)
__device__ __constant__ float kS    = 64.0f;
__device__ __constant__ float kCosM = 0.8775825618903728f;   // cos(0.5)
__device__ __constant__ float kSinM = 0.47942553860420301f;  // sin(0.5)
__device__ __constant__ float kTH   = -0.8775825618903728f;  // cos(pi-0.5)
__device__ __constant__ float kMM   = 0.23971276930210151f;  // sin(pi-0.5)*0.5
__device__ __constant__ float kEps  = 1e-7f;

// ---------------------------------------------------------------------------
// Kernel 1: row-normalize weight [C,E] (f32) and emit bf16. One wave per row.
// Memory-bound: 153.6 MB read + 76.8 MB write. The only full-f32 pass.
// ---------------------------------------------------------------------------
__global__ __launch_bounds__(256) void arcface_norm_w(const float* __restrict__ W,
                                                      bf16_t* __restrict__ Wn) {
    const int wid  = threadIdx.x >> 5;
    const int lane = threadIdx.x & 31;
    const int row  = blockIdx.x * 8 + wid;
    if (row >= NUM_CLASSES) return;

    const float* src = W + (size_t)row * EMBED;
    float v[12];
    float ss = 0.0f;
#pragma unroll
    for (int j = 0; j < 12; ++j) {
        v[j] = src[lane + 32 * j];
        ss += v[j] * v[j];
    }
#pragma unroll
    for (int off = 16; off >= 1; off >>= 1)
        ss += __shfl_xor(ss, off, 32);

    const float scale = rsqrtf(ss);
    bf16_t* dst = Wn + (size_t)row * EMBED;
#pragma unroll
    for (int j = 0; j < 12; ++j)
        dst[lane + 32 * j] = (bf16_t)(v[j] * scale);
}

// ---------------------------------------------------------------------------
// Kernel 2: embeddings f32 -> bf16 (NOT normalized, per reference). Tiny.
// ---------------------------------------------------------------------------
__global__ __launch_bounds__(256) void arcface_cvt_emb(const float* __restrict__ E,
                                                       bf16_t* __restrict__ Eb) {
    const int idx = blockIdx.x * 256 + threadIdx.x;
    if (idx < BATCH * EMBED) Eb[idx] = (bf16_t)E[idx];
}

// ---------------------------------------------------------------------------
// Kernel 3: fused bf16 WMMA GEMM + clip + ArcFace margin + exp partial-sums.
// Block = 8 waves; block owns one N-tile (16 classes) x all 512 batch rows.
// The 12 KB B tile is staged Global->LDS with the gfx1250 async copy engine
// (ASYNCcnt), then consumed via ds_load_b128; WMMAs only wait on A loads.
// ---------------------------------------------------------------------------
__global__ __launch_bounds__(256) void arcface_gemm(const bf16_t* __restrict__ Eb,
                                                    const bf16_t* __restrict__ Wn,
                                                    const long long* __restrict__ labels,
                                                    float* __restrict__ partial,  // [NTILES][BATCH]
                                                    float* __restrict__ tlogit)   // [BATCH]
{
    __shared__ __align__(16) bf16_t smemB[16 * BROW_PAD];

    const int tid    = threadIdx.x;
    const int wid    = tid >> 5;
    const int lane   = tid & 31;
    const int c0     = blockIdx.x * 16;
    const int mbase0 = wid * 64;
    const int col    = c0 + (lane & 15);
    const int aoff   = (lane < 16) ? 0 : 8;    // A: upper half-wave holds K+8
    const int boff   = (lane < 16) ? 0 : 8;    // B: upper half-wave holds K+8

    // ---- async stage of the 16x384 bf16 weight tile into LDS --------------
    // 768 chunks of 16 B; each of the 256 threads issues 3 async b128 copies.
#pragma unroll
    for (int t = 0; t < 3; ++t) {
        const int chunk = tid + t * 256;          // 0..767
        const int row   = chunk / 48;             // class row within tile
        const int cb    = chunk % 48;             // 16-byte column chunk
        const bf16_t* gsrc = Wn + (size_t)(c0 + row) * EMBED + cb * 8;
        // low 32 bits of a generic LDS address are the raw LDS offset
        const uint32_t ldsoff = (uint32_t)(uintptr_t)(smemB + row * BROW_PAD + cb * 8);
        const uint64_t gaddr  = (uint64_t)(uintptr_t)gsrc;
        asm volatile("global_load_async_to_lds_b128 %0, %1, off"
                     :: "v"(ldsoff), "v"(gaddr)
                     : "memory");
    }
    asm volatile("s_wait_asynccnt 0" ::: "memory");
    __syncthreads();

    // ---- K-loop: B fragments from LDS, A fragments from L2-resident emb ---
    const v8f vzero = {};
    v8f acc[4];
#pragma unroll
    for (int mt = 0; mt < 4; ++mt) acc[mt] = vzero;

    const bf16_t* bsrcRow = smemB + (size_t)(lane & 15) * BROW_PAD + boff;

    for (int k0 = 0; k0 < EMBED; k0 += 32) {
        v16bf bfrag;
        {
            const bf16_t* bsrc = bsrcRow + k0;
            v8bf blo = *(const v8bf*)bsrc;         // ds_load_b128
            v8bf bhi = *(const v8bf*)(bsrc + 16);  // ds_load_b128
#pragma unroll
            for (int i = 0; i < 8; ++i) { bfrag[i] = blo[i]; bfrag[8 + i] = bhi[i]; }
        }

#pragma unroll
        for (int mt = 0; mt < 4; ++mt) {
            const bf16_t* arow =
                Eb + (size_t)(mbase0 + mt * 16 + (lane & 15)) * EMBED + k0 + aoff;
            if (mt == 0 && k0 + 32 < EMBED)
                __builtin_prefetch(arow + 32, 0, 1);   // global_prefetch_b8
            v8bf alo = *(const v8bf*)(arow);
            v8bf ahi = *(const v8bf*)(arow + 16);
            v16bf afrag;
#pragma unroll
            for (int i = 0; i < 8; ++i) { afrag[i] = alo[i]; afrag[8 + i] = ahi[i]; }

            acc[mt] = __builtin_amdgcn_wmma_f32_16x16x32_bf16(
                false, afrag, false, bfrag, (short)0, acc[mt], false, false);
        }
    }

    // Epilogue: C/D layout -> VGPR v, lanes 0-15: (M=v, N=lane); lanes 16-31: (M=v+8, N=lane-16)
#pragma unroll
    for (int mt = 0; mt < 4; ++mt) {
#pragma unroll
        for (int v = 0; v < 8; ++v) {
            const int m = mbase0 + mt * 16 + v + ((lane >= 16) ? 8 : 0);
            float cosv = acc[mt][v];
            cosv = fminf(fmaxf(cosv, -1.0f + kEps), 1.0f - kEps);

            const long long lbl = labels[m];
            const bool is_t = (lbl == (long long)col);

            const float sinv = sqrtf(1.0f - cosv * cosv);
            const float cm   = cosv * kCosM - sinv * kSinM;
            const float adjT = (cosv > kTH) ? cm : (cosv - kMM);
            const float adj  = is_t ? adjT : cosv;
            const float t    = kS * adj;
            if (is_t) tlogit[m] = t;

            float e = __expf(t - kS);   // all logits <= S, so S is a safe max
#pragma unroll
            for (int off = 1; off < 16; off <<= 1)
                e += __shfl_xor(e, off, 32);   // sum over the tile's 16 classes
            if ((lane & 15) == 0)
                partial[(size_t)blockIdx.x * BATCH + m] = e;
        }
    }
}

// ---------------------------------------------------------------------------
// Kernel 4: per-row logsumexp assembly + mean NLL. One block, 512 threads.
// ---------------------------------------------------------------------------
__global__ __launch_bounds__(512) void arcface_final(const float* __restrict__ partial,
                                                     const float* __restrict__ tlogit,
                                                     float* __restrict__ out) {
    __shared__ float red[512];
    const int m = threadIdx.x;
    float s = 0.0f;
    for (int b = 0; b < NTILES; ++b)
        s += partial[(size_t)b * BATCH + m];
    const float lse = kS + __logf(s);
    red[m] = lse - tlogit[m];
    __syncthreads();
    for (int off = 256; off >= 1; off >>= 1) {
        if (m < off) red[m] += red[m + off];
        __syncthreads();
    }
    if (m == 0) out[0] = red[0] / (float)BATCH;
}

// ---------------------------------------------------------------------------
extern "C" void kernel_launch(void* const* d_in, const int* in_sizes, int n_in,
                              void* d_out, int out_size, void* d_ws, size_t ws_size,
                              hipStream_t stream) {
    const float*     emb    = (const float*)d_in[0];       // [512,384] f32
    const long long* labels = (const long long*)d_in[1];   // [512] int64
    const float*     W      = (const float*)d_in[2];       // [100000,384] f32
    float*           out    = (float*)d_out;               // scalar f32

    char*  ws  = (char*)d_ws;
    size_t off = 0;
    bf16_t* Wn = (bf16_t*)(ws + off); off += (size_t)NUM_CLASSES * EMBED * sizeof(bf16_t);
    bf16_t* Eb = (bf16_t*)(ws + off); off += (size_t)BATCH * EMBED * sizeof(bf16_t);
    off = (off + 255) & ~(size_t)255;
    float* tlogit = (float*)(ws + off); off += (size_t)BATCH * sizeof(float);
    off = (off + 255) & ~(size_t)255;
    float* partial = (float*)(ws + off); off += (size_t)NTILES * BATCH * sizeof(float);
    (void)ws_size; (void)in_sizes; (void)n_in; (void)out_size;

    arcface_norm_w<<<NUM_CLASSES / 8, 256, 0, stream>>>(W, Wn);
    arcface_cvt_emb<<<(BATCH * EMBED) / 256, 256, 0, stream>>>(emb, Eb);
    arcface_gemm<<<NTILES, 256, 0, stream>>>(Eb, Wn, labels, partial, tlogit);
    arcface_final<<<1, 512, 0, stream>>>(partial, tlogit, out);
}
